// BiomechanicalConstraintModule_66365834657797
// MI455X (gfx1250) — compile-verified, matrix operand order checked by reading, not run on hardware
//
#include <hip/hip_runtime.h>
#include <hip/hip_bf16.h>
#include <math.h>

// ---------------------------------------------------------------------------
// Problem constants (from reference): B=64, T=243, J=17, C=128
// ---------------------------------------------------------------------------
#define B_   64
#define T_   243
#define J_   17
#define C_   128
#define TT   8                 // time steps per workgroup tile
#define NTILES_T 31            // ceil(243/8)
#define FROWS 144              // TT*J=136 rows padded to 9 WMMA M-tiles
#define FSTRIDE 136            // 128 + 8 pad (bank-conflict-free LDS stride)

// output offsets (floats), in reference return order
#define OUT_PANG   0                                   // [B,T,17,3] = 793152
#define OUT_AANG   (793152)                            // [B,T,17,1] = 264384
#define OUT_PBONE  (793152 + 264384)                   // [B,T,15,1] = 233280
#define OUT_ABONE  (793152 + 264384 + 233280)          // [B,T,15,1] = 233280
#define OUT_LOGIT  (793152 + 264384 + 233280 + 233280) // [B,8] = 512

// small-weight LDS layout
#define SW_AB1 0      // 64
#define SW_AW2 64     // 192
#define SW_AB2 256    // 3
#define SW_BB1 259    // 128
#define SW_BW2 387    // 128
#define SW_BB2 515    // 1
#define SW_TOT 516

__device__ const int cCONN_A[15] = {0,1,2,0,4,5,0,7,8,0,10,11,0,13,14};
__device__ const int cCONN_B[15] = {1,2,3,4,5,6,7,8,9,10,11,12,13,14,15};

typedef __attribute__((ext_vector_type(16))) __bf16 v16bf;
typedef __attribute__((ext_vector_type(8)))  float  v8f;

union FragU { v16bf v; uint4 q[2]; };

__device__ __forceinline__ unsigned short f2bf(float f) {
  unsigned u = __float_as_uint(f);
  unsigned r = (u + 0x7fffu + ((u >> 16) & 1u)) >> 16;   // RNE truncate
  return (unsigned short)r;
}

// Branch-free erf (Abramowitz-Stegun 7.1.26, |err| <= 1.5e-7):
// keeps the WMMA loop free of EXEC-modifying compares; rcp/exp are TRANS ops
// that co-execute with the matrix pipe.
__device__ __forceinline__ float erf_approx(float x) {
  float ax = fabsf(x);
  float t  = __builtin_amdgcn_rcpf(1.0f + 0.3275911f * ax);
  float p  = fmaf(1.061405429f, t, -1.453152027f);
  p = fmaf(p, t, 1.421413741f);
  p = fmaf(p, t, -0.284496736f);
  p = fmaf(p, t, 0.254829592f);
  p = p * t;
  float e  = __expf(-ax * ax);
  float r  = fmaf(-p, e, 1.0f);
  return copysignf(r, x);
}

__device__ __forceinline__ float gelu_exact(float x) {
  return 0.5f * x * (1.0f + erf_approx(x * 0.70710678118654752440f));
}

// ---------------------------------------------------------------------------
// Prologue: zero the action-feature accumulator and repack aw1/bw1 into the
// per-lane WMMA B-fragment layout (bf16).  Fragment element e of lane L holds
// B[k][n] with n = nt*16 + (L&15), k = kt*32 + 8*(L>>4) + (e&7) + (e&8 ? 16:0)
// ---------------------------------------------------------------------------
__global__ void prep_kernel(const float* __restrict__ aw1,   // [128,64]
                            const float* __restrict__ bw1,   // [256,128]
                            float* __restrict__ accum,       // [64,2176]
                            unsigned short* __restrict__ aw1pack, // 16 frags * 512
                            unsigned short* __restrict__ bw1pack) // 64 frags * 512
{
  int tid = blockIdx.x * blockDim.x + threadIdx.x;
  if (tid < 64 * 2176) accum[tid] = 0.0f;
  if (tid < 16 * 512) {
    int f = tid >> 9, lane = (tid >> 4) & 31, e = tid & 15;
    int kt = f >> 2, nt = f & 3;
    int n = nt * 16 + (lane & 15);
    int k = kt * 32 + ((lane >> 4) << 3) + (e & 7) + ((e & 8) ? 16 : 0);
    aw1pack[tid] = f2bf(aw1[k * 64 + n]);
  }
  if (tid < 64 * 512) {
    int f = tid >> 9, lane = (tid >> 4) & 31, e = tid & 15;
    int kt = f >> 3, nt = f & 7;
    int n = nt * 16 + (lane & 15);
    int k = kt * 32 + ((lane >> 4) << 3) + (e & 7) + ((e & 8) ? 16 : 0);
    bw1pack[tid] = f2bf(bw1[k * 128 + n]);
  }
}

// ---------------------------------------------------------------------------
// Fused main kernel: one WG per (b, 8-timestep tile).
//  stage 1: load features tile -> LDS (bf16) + fp32 time-sum atomics
//  stage 2: angle MLP  (WMMA 16x16x32 bf16, K=128, N=64) + GELU + N=3 head
//  stage 3: bone  MLP  (WMMA, K=256 via LDS row-gather,  N=128) + GELU + N=1
// ---------------------------------------------------------------------------
__global__ __launch_bounds__(256) void fused_mlp_kernel(
    const float* __restrict__ features,
    const float* __restrict__ ab1, const float* __restrict__ aw2,
    const float* __restrict__ ab2, const float* __restrict__ bb1,
    const float* __restrict__ bw2, const float* __restrict__ bb2,
    const unsigned short* __restrict__ aw1pack,
    const unsigned short* __restrict__ bw1pack,
    float* __restrict__ accum,
    float* __restrict__ outAngles,    // [B,T,17,3]
    float* __restrict__ outBoneLen)   // [B,T,15]
{
  __shared__ __align__(16) unsigned short featLDS[FROWS * FSTRIDE];
  __shared__ float smallW[SW_TOT];

  const int b    = blockIdx.y;
  const int t0   = blockIdx.x * TT;
  const int tid  = threadIdx.x;
  const int lane = tid & 31;
  const int wave = tid >> 5;

  // zero LDS tile (covers M padding + partial-t rows)
  for (int i = tid; i < (FROWS * FSTRIDE) / 8; i += 256)
    ((uint4*)featLDS)[i] = make_uint4(0u, 0u, 0u, 0u);

  // stage small weights
  for (int i = tid; i < SW_TOT; i += 256) {
    float v;
    if      (i < 64)  v = ab1[i];
    else if (i < 256) v = aw2[i - SW_AW2];
    else if (i < 259) v = ab2[i - SW_AB2];
    else if (i < 387) v = bb1[i - SW_BB1];
    else if (i < 515) v = bw2[i - SW_BW2];
    else              v = bb2[0];
    smallW[i] = v;
  }
  __syncthreads();

  // ---- stage 1: load features (fp32 -> bf16 LDS) + time-sum atomics ----
  for (int p = tid; p < J_ * C_; p += 256) {
    int j = p >> 7, c = p & 127;
    float s = 0.0f;
    const float* src = features + (((size_t)(b * T_ + t0)) * J_ + j) * C_ + c;
    #pragma unroll
    for (int tl = 0; tl < TT; ++tl) {
      if (t0 + tl < T_) {
        float f = src[(size_t)tl * J_ * C_];
        s += f;
        featLDS[(tl * J_ + j) * FSTRIDE + c] = f2bf(f);
      }
    }
    atomicAdd(&accum[b * (J_ * C_) + p], s);
  }
  // hint next t-tile into cache while we compute
  if (t0 + TT < T_)
    __builtin_prefetch(features + (((size_t)(b * T_ + t0 + TT)) * J_) * C_ + tid, 0, 0);
  __syncthreads();

  const int m   = lane & 15;
  const int kb  = (lane >> 4) << 3;
  const int nlo = lane & 15;

  // ---- stage 2: angle MLP -------------------------------------------------
  for (int mt = wave; mt < 9; mt += 8) {
    v8f acc[4];
    #pragma unroll
    for (int i = 0; i < 4; ++i) acc[i] = (v8f){0.f,0.f,0.f,0.f,0.f,0.f,0.f,0.f};

    #pragma unroll
    for (int kt = 0; kt < 4; ++kt) {
      FragU a;
      const uint4* pa = (const uint4*)(featLDS + (mt * 16 + m) * FSTRIDE + kt * 32 + kb);
      a.q[0] = pa[0];
      a.q[1] = pa[2];          // +16 bf16 = +2 uint4
      #pragma unroll
      for (int nt = 0; nt < 4; ++nt) {
        FragU bq;
        const uint4* pb = (const uint4*)(aw1pack + ((kt * 4 + nt) * 512 + lane * 16));
        bq.q[0] = pb[0]; bq.q[1] = pb[1];
        acc[nt] = __builtin_amdgcn_wmma_f32_16x16x32_bf16(
            false, a.v, false, bq.v, (short)0, acc[nt], false, false);
      }
    }

    // bias + GELU, then tiny [64 -> 3] head via half-wave shuffle reductions
    float g[4][8];
    #pragma unroll
    for (int nt = 0; nt < 4; ++nt) {
      float bias = smallW[SW_AB1 + nt * 16 + nlo];
      #pragma unroll
      for (int r = 0; r < 8; ++r) g[nt][r] = gelu_exact(acc[nt][r] + bias);
    }
    #pragma unroll
    for (int r = 0; r < 8; ++r) {
      float s0 = 0.f, s1 = 0.f, s2 = 0.f;
      #pragma unroll
      for (int nt = 0; nt < 4; ++nt) {
        int n = nt * 16 + nlo;
        float gv = g[nt][r];
        s0 += gv * smallW[SW_AW2 + n * 3 + 0];
        s1 += gv * smallW[SW_AW2 + n * 3 + 1];
        s2 += gv * smallW[SW_AW2 + n * 3 + 2];
      }
      #pragma unroll
      for (int off = 8; off >= 1; off >>= 1) {
        s0 += __shfl_xor(s0, off, 32);
        s1 += __shfl_xor(s1, off, 32);
        s2 += __shfl_xor(s2, off, 32);
      }
      if (nlo == 0) {
        int row = mt * 16 + r + ((lane >> 4) << 3);
        if (row < TT * J_) {
          int tl = row / J_, joint = row - tl * J_;
          int t = t0 + tl;
          if (t < T_) {
            float* o = outAngles + (((size_t)(b * T_ + t)) * J_ + joint) * 3;
            o[0] = s0 + smallW[SW_AB2 + 0];
            o[1] = s1 + smallW[SW_AB2 + 1];
            o[2] = s2 + smallW[SW_AB2 + 2];
          }
        }
      }
    }
  }

  // ---- stage 3: bone MLP (K=256 = concat of two gathered joint rows) ------
  {
    const int mt = wave;                     // 8 waves <-> 8 bone M-tiles
    v8f acc[8];
    #pragma unroll
    for (int i = 0; i < 8; ++i) acc[i] = (v8f){0.f,0.f,0.f,0.f,0.f,0.f,0.f,0.f};

    int rb = mt * 16 + m;
    int rowA, rowB;
    if (rb < TT * 15) {
      int tl = rb / 15, bn = rb - tl * 15;
      rowA = tl * J_ + cCONN_A[bn];
      rowB = tl * J_ + cCONN_B[bn];
    } else {
      rowA = FROWS - 1; rowB = FROWS - 1;    // guaranteed-zero padded row
    }

    #pragma unroll
    for (int kt = 0; kt < 8; ++kt) {
      int k1 = kt * 32 + kb;
      int k2 = k1 + 16;
      int r1 = (k1 < C_) ? rowA : rowB, c1 = k1 & (C_ - 1);
      int r2 = (k2 < C_) ? rowA : rowB, c2 = k2 & (C_ - 1);
      FragU a;
      a.q[0] = *(const uint4*)(featLDS + r1 * FSTRIDE + c1);
      a.q[1] = *(const uint4*)(featLDS + r2 * FSTRIDE + c2);
      #pragma unroll
      for (int nt = 0; nt < 8; ++nt) {
        FragU bq;
        const uint4* pb = (const uint4*)(bw1pack + ((kt * 8 + nt) * 512 + lane * 16));
        bq.q[0] = pb[0]; bq.q[1] = pb[1];
        acc[nt] = __builtin_amdgcn_wmma_f32_16x16x32_bf16(
            false, a.v, false, bq.v, (short)0, acc[nt], false, false);
      }
    }

    #pragma unroll
    for (int r = 0; r < 8; ++r) {
      float s = 0.f;
      #pragma unroll
      for (int nt = 0; nt < 8; ++nt) {
        int n = nt * 16 + nlo;
        float gv = gelu_exact(acc[nt][r] + smallW[SW_BB1 + n]);
        s += gv * smallW[SW_BW2 + n];
      }
      #pragma unroll
      for (int off = 8; off >= 1; off >>= 1) s += __shfl_xor(s, off, 32);
      if (nlo == 0) {
        int rr = mt * 16 + r + ((lane >> 4) << 3);
        if (rr < TT * 15) {
          int tl = rr / 15, bn = rr - tl * 15;
          int t = t0 + tl;
          if (t < T_) {
            float v = s + smallW[SW_BB2];
            outBoneLen[((size_t)(b * T_ + t)) * 15 + bn] = v > 0.f ? v : 0.f;
          }
        }
      }
    }
  }
}

// ---------------------------------------------------------------------------
// Geometry from pose3d: actual bone lengths + actual joint angles
// ---------------------------------------------------------------------------
__global__ void pose_kernel(const float* __restrict__ pose3d,
                            float* __restrict__ outActAng,   // [B,T,17]
                            float* __restrict__ outActLen)   // [B,T,15]
{
  int idx = blockIdx.x * blockDim.x + threadIdx.x;   // b*T + t
  if (idx >= B_ * T_) return;
  const float* p = pose3d + (size_t)idx * J_ * 3;

  #pragma unroll
  for (int i = 0; i < 15; ++i) {
    int a = cCONN_A[i], c = cCONN_B[i];
    float dx = p[c*3+0]-p[a*3+0], dy = p[c*3+1]-p[a*3+1], dz = p[c*3+2]-p[a*3+2];
    outActLen[(size_t)idx * 15 + i] = sqrtf(dx*dx + dy*dy + dz*dz);
  }

  float* oa = outActAng + (size_t)idx * J_;
  #pragma unroll
  for (int j = 0; j < J_; ++j) oa[j] = 0.0f;

  const int AJ[4] = {11,14,5,8}, AP[4] = {10,13,4,7}, AC[4] = {12,15,6,9};
  #pragma unroll
  for (int i = 0; i < 4; ++i) {
    int jj = AJ[i], pp = AP[i], cc = AC[i];
    float v1x = p[jj*3+0]-p[pp*3+0], v1y = p[jj*3+1]-p[pp*3+1], v1z = p[jj*3+2]-p[pp*3+2];
    float v2x = p[cc*3+0]-p[jj*3+0], v2y = p[cc*3+1]-p[jj*3+1], v2z = p[cc*3+2]-p[jj*3+2];
    float n1 = sqrtf(v1x*v1x + v1y*v1y + v1z*v1z) + 1e-10f;
    float n2 = sqrtf(v2x*v2x + v2y*v2y + v2z*v2z) + 1e-10f;
    float dot = (v1x*v2x + v1y*v2y + v1z*v2z) / (n1 * n2);
    dot = fminf(fmaxf(dot, -1.0f + 1e-7f), 1.0f - 1e-7f);
    oa[jj] = acosf(dot);
  }
}

// ---------------------------------------------------------------------------
// Action head: hidden = GELU(mean_feat @ cw1 + cb1), one wave per (row, n)
// ---------------------------------------------------------------------------
__global__ void action_hidden_kernel(const float* __restrict__ accum,  // [64,2176] time-sums
                                     const float* __restrict__ cw1,    // [2176,64]
                                     const float* __restrict__ cb1,    // [64]
                                     float* __restrict__ hbuf)         // [64,64]
{
  int wid  = (blockIdx.x * blockDim.x + threadIdx.x) >> 5;  // 4096 waves
  int lane = threadIdx.x & 31;
  int row = wid >> 6, n = wid & 63;
  const float invT = 1.0f / (float)T_;
  float s = 0.0f;
  for (int k = lane; k < J_ * C_; k += 32)
    s += accum[row * (J_ * C_) + k] * invT * cw1[k * 64 + n];
  #pragma unroll
  for (int off = 16; off >= 1; off >>= 1) s += __shfl_xor(s, off, 32);
  if (lane == 0) hbuf[row * 64 + n] = gelu_exact(s + cb1[n]);
}

__global__ void action_logits_kernel(const float* __restrict__ hbuf,
                                     const float* __restrict__ cw2,   // [64,8]
                                     const float* __restrict__ cb2,   // [8]
                                     float* __restrict__ outLogits)   // [64,8]
{
  int tid = blockIdx.x * blockDim.x + threadIdx.x;
  if (tid >= 64 * 8) return;
  int row = tid >> 3, j = tid & 7;
  float s = 0.0f;
  #pragma unroll
  for (int k = 0; k < 64; ++k) s += hbuf[row * 64 + k] * cw2[k * 8 + j];
  outLogits[tid] = s + cb2[j];
}

// ---------------------------------------------------------------------------
extern "C" void kernel_launch(void* const* d_in, const int* in_sizes, int n_in,
                              void* d_out, int out_size, void* d_ws, size_t ws_size,
                              hipStream_t stream) {
  const float* features = (const float*)d_in[0];
  const float* pose3d   = (const float*)d_in[1];
  const float* aw1 = (const float*)d_in[2];
  const float* ab1 = (const float*)d_in[3];
  const float* aw2 = (const float*)d_in[4];
  const float* ab2 = (const float*)d_in[5];
  const float* bw1 = (const float*)d_in[6];
  const float* bb1 = (const float*)d_in[7];
  const float* bw2 = (const float*)d_in[8];
  const float* bb2 = (const float*)d_in[9];
  const float* cw1 = (const float*)d_in[10];
  const float* cb1 = (const float*)d_in[11];
  const float* cw2 = (const float*)d_in[12];
  const float* cb2 = (const float*)d_in[13];

  float* out = (float*)d_out;
  float* outAngles  = out + OUT_PANG;
  float* outActAng  = out + OUT_AANG;
  float* outBoneLen = out + OUT_PBONE;
  float* outActLen  = out + OUT_ABONE;
  float* outLogits  = out + OUT_LOGIT;

  // workspace layout
  float* accum = (float*)d_ws;                       // 64*2176 f32
  float* hbuf  = accum + 64 * (J_ * C_);             // 64*64  f32
  unsigned short* aw1pack = (unsigned short*)(hbuf + 64 * 64);  // 16*512 bf16
  unsigned short* bw1pack = aw1pack + 16 * 512;                 // 64*512 bf16

  prep_kernel<<<544, 256, 0, stream>>>(aw1, bw1, accum, aw1pack, bw1pack);

  dim3 grid(NTILES_T, B_);
  fused_mlp_kernel<<<grid, 256, 0, stream>>>(features, ab1, aw2, ab2, bb1, bw2,
                                             bb2, aw1pack, bw1pack, accum,
                                             outAngles, outBoneLen);

  pose_kernel<<<(B_ * T_ + 255) / 256, 256, 0, stream>>>(pose3d, outActAng, outActLen);

  action_hidden_kernel<<<(64 * 64 * 32) / 256, 256, 0, stream>>>(accum, cw1, cb1, hbuf);
  action_logits_kernel<<<2, 256, 0, stream>>>(hbuf, cw2, cb2, outLogits);
}
